// CodecAttention_34308198761010
// MI455X (gfx1250) — matrix operand order repaired and checked by reading, not compiled, not run on hardware
//
#include <hip/hip_runtime.h>
#include <hip/hip_bf16.h>

// ---------------- types ----------------
typedef __bf16 bf16x16 __attribute__((ext_vector_type(16)));
typedef __bf16 bf16x8  __attribute__((ext_vector_type(8)));
typedef float  f32x8   __attribute__((ext_vector_type(8)));
typedef int    i32x4   __attribute__((ext_vector_type(4)));

// ---------------- problem constants (from reference setup_inputs) ----------------
#define BATCH 2
#define SEQ   2048
#define MODEL 1024
#define HEADS 16
#define KVH   4
#define HDIM  64
#define WIN   512
#define QKVN  (HEADS*HDIM + 2*KVH*HDIM)   // 1536
#define ROWS  (BATCH*SEQ)                 // 4096

// ---------------- CDNA5 async global->LDS copy (ASYNCcnt path) ----------------
#if __has_builtin(__builtin_amdgcn_global_load_async_to_lds_b128)
#define HAVE_ASYNC_LDS 1
#endif

__device__ __forceinline__ void async_copy16(const void* g, void* l) {
#ifdef HAVE_ASYNC_LDS
    __builtin_amdgcn_global_load_async_to_lds_b128(
        (__attribute__((address_space(1))) i32x4*)g,
        (__attribute__((address_space(3))) i32x4*)l, 0, 0);
#else
    *(uint4*)l = *(const uint4*)g;
#endif
}

__device__ __forceinline__ void async_wait0() {
#ifdef HAVE_ASYNC_LDS
#if __has_builtin(__builtin_amdgcn_s_wait_asynccnt)
    __builtin_amdgcn_s_wait_asynccnt(0);
#else
    asm volatile("s_wait_asynccnt 0" ::: "memory");
#endif
#endif
}

// ---------------- WMMA helpers ----------------
__device__ __forceinline__ f32x8 wmma_bf16(bf16x16 a, bf16x16 b, f32x8 c) {
    return __builtin_amdgcn_wmma_f32_16x16x32_bf16(false, a, false, b, (short)0, c, false, false);
}

// A-matrix 16x32 bf16 fragment. Lane L: row m = L%16.
// elements 0..7  -> K = k0 + (L/16)*8 + e
// elements 8..15 -> K = k0 + 16 + (L/16)*8 + (e-8)
__device__ __forceinline__ bf16x16 load_a_frag(const __bf16* base, int row0, int k0, int ld) {
    int lane = threadIdx.x & 31;
    int m = lane & 15, half = lane >> 4;
    const __bf16* p = base + (size_t)(row0 + m) * ld + k0 + half * 8;
    bf16x8 lo = *(const bf16x8*)(p);
    bf16x8 hi = *(const bf16x8*)(p + 16);
    return __builtin_shufflevector(lo, hi, 0,1,2,3,4,5,6,7,8,9,10,11,12,13,14,15);
}

// B-matrix 32x16 bf16 fragment for C[i,n] = sum_k A[i,k]*W[n,k] (W row-major [N,K]).
// Lane L: col n = L%16, elements 0..7 -> K = k0 + (L/16)*8 + e, elements 8..15 -> K = k0 + 16 + (L/16)*8 + (e-8).
__device__ __forceinline__ bf16x16 load_b_frag(const __bf16* base, int col0, int k0, int ld) {
    int lane = threadIdx.x & 31;
    int n = lane & 15, half = lane >> 4;
    const __bf16* p = base + (size_t)(col0 + n) * ld + k0 + half * 8;
    bf16x8 lo = *(const bf16x8*)(p);
    bf16x8 hi = *(const bf16x8*)(p + 16);
    return __builtin_shufflevector(lo, hi, 0,1,2,3,4,5,6,7,8,9,10,11,12,13,14,15);
}

// ---------------- kernel 1: f32 -> bf16 convert ----------------
__global__ void cvt_f32_bf16(const float* __restrict__ src, __bf16* __restrict__ dst, int n) {
    int i = blockIdx.x * blockDim.x + threadIdx.x;
    if (i < n) dst[i] = (__bf16)src[i];
}

// ---------------- kernel 2/5: bf16 WMMA GEMM with double-buffered async LDS staging ----
// C[i,n] = sum_k A[i,k] * W[n,k]. Block = 256 threads = 8 waves (2 x 4).
// Block tile = 64 rows x 256 cols, wave tile = 32 x 64, k-step 32.
#define BM 64
#define BN 256
#define BK 32
__global__ __launch_bounds__(256) void gemm_bf16(const __bf16* __restrict__ A,
                                                 const __bf16* __restrict__ W,
                                                 float* __restrict__ C,
                                                 int N, int K) {
    __shared__ __attribute__((aligned(128))) __bf16 As[2][BM * BK];   // 2 x 4 KB
    __shared__ __attribute__((aligned(128))) __bf16 Bs[2][BN * BK];   // 2 x 16 KB

    int t = threadIdx.x;
    int lane = t & 31, wid = t >> 5;
    int wrow = wid >> 2, wcol = wid & 3;
    int rowBase = blockIdx.y * BM;
    int colBase = blockIdx.x * BN;

    auto stage = [&](int buf, int kk) {
        {   // A tile 64x32: one 16B chunk per thread
            int row = t >> 2, off = (t & 3) * 8;
            async_copy16(A + (size_t)(rowBase + row) * K + kk + off,
                         &As[buf][row * BK + off]);
        }
        #pragma unroll
        for (int i = 0; i < 4; i++) {   // B tile 256x32: four 16B chunks per thread
            int e = t + i * 256;
            int row = e >> 2, off = (e & 3) * 8;
            async_copy16(W + (size_t)(colBase + row) * K + kk + off,
                         &Bs[buf][row * BK + off]);
        }
    };

    f32x8 acc[2][4] = {};
    int KT = K / BK;
    stage(0, 0);
    for (int kt = 0; kt < KT; kt++) {
        int cur = kt & 1;
        async_wait0();          // this wave's DMA into `cur` done
        __syncthreads();        // everyone's DMA done; everyone done reading 1^cur
        if (kt + 1 < KT) stage(cur ^ 1, (kt + 1) * BK);   // overlap with compute

        bf16x16 a0 = load_a_frag(&As[cur][0], wrow * 32,      0, BK);
        bf16x16 a1 = load_a_frag(&As[cur][0], wrow * 32 + 16, 0, BK);
        bf16x16 b[4];
        #pragma unroll
        for (int j = 0; j < 4; j++)
            b[j] = load_b_frag(&Bs[cur][0], wcol * 64 + j * 16, 0, BK);
        #pragma unroll
        for (int j = 0; j < 4; j++) {
            acc[0][j] = wmma_bf16(a0, b[j], acc[0][j]);
            acc[1][j] = wmma_bf16(a1, b[j], acc[1][j]);
        }
    }

    int n = lane & 15, half = lane >> 4;
    #pragma unroll
    for (int i = 0; i < 2; i++)
        #pragma unroll
        for (int j = 0; j < 4; j++)
            #pragma unroll
            for (int r = 0; r < 8; r++) {
                int rr = rowBase + wrow * 32 + i * 16 + r + 8 * half;
                int cc = colBase + wcol * 64 + j * 16 + n;
                C[(size_t)rr * N + cc] = acc[i][j][r];
            }
}

// ---------------- kernel 3: RMSNorm + relayout ----------------
__global__ __launch_bounds__(256) void norm_relayout(const float* __restrict__ QKV,
                                                     const float* __restrict__ qw,
                                                     const float* __restrict__ kw,
                                                     __bf16* __restrict__ Qb,
                                                     __bf16* __restrict__ Kb,
                                                     __bf16* __restrict__ Vt) {
    __shared__ float red[256];
    int token = blockIdx.x;
    int b = token / SEQ, s = token % SEQ;
    int t = threadIdx.x;
    const float* row = QKV + (size_t)token * QKVN;

    float ss = 0.f;
    for (int i = t; i < HEADS*HDIM; i += 256) { float x = row[i]; ss += x * x; }
    red[t] = ss; __syncthreads();
    for (int o = 128; o > 0; o >>= 1) { if (t < o) red[t] += red[t + o]; __syncthreads(); }
    float qscale = rsqrtf(red[0] / (float)(HEADS*HDIM) + 1e-6f);
    __syncthreads();

    float kx = row[HEADS*HDIM + t];
    red[t] = kx * kx; __syncthreads();
    for (int o = 128; o > 0; o >>= 1) { if (t < o) red[t] += red[t + o]; __syncthreads(); }
    float kscale = rsqrtf(red[0] / (float)(KVH*HDIM) + 1e-6f);

    for (int i = t; i < HEADS*HDIM; i += 256) {
        int h = i >> 6, d = i & 63;
        Qb[(((size_t)b*HEADS + h) * SEQ + s) * HDIM + d] = (__bf16)(row[i] * qscale * qw[i]);
    }
    {
        int kv = t >> 6, d = t & 63;
        Kb[(((size_t)b*KVH + kv) * SEQ + s) * HDIM + d] = (__bf16)(kx * kscale * kw[t]);
    }
    {
        int kv = t >> 6, d = t & 63;
        Vt[(((size_t)b*KVH + kv) * HDIM + d) * SEQ + s] = (__bf16)(row[HEADS*HDIM + KVH*HDIM + t]);
    }
}

// ---------------- kernel 4: sliding-window ALiBi flash attention ----------------
__global__ __launch_bounds__(256) void attn_kernel(const __bf16* __restrict__ Qb,
                                                   const __bf16* __restrict__ Kb,
                                                   const __bf16* __restrict__ Vt,
                                                   __bf16* __restrict__ Ob) {
    __shared__ __attribute__((aligned(64))) __bf16 plds[8][16 * 32];
    int lane = threadIdx.x & 31;
    int wid  = threadIdx.x >> 5;
    int gw = blockIdx.x * 8 + wid;
    int qt = gw & 127;
    int h  = (gw >> 7) & (HEADS - 1);
    int b  = gw >> 11;
    int q0 = qt * 16;
    int kv = h >> 2;

    const __bf16* Qh = Qb + (((size_t)b*HEADS + h) * SEQ) * HDIM;
    const __bf16* Kh = Kb + (((size_t)b*KVH + kv) * SEQ) * HDIM;
    const __bf16* Vh = Vt + (((size_t)b*KVH + kv) * HDIM) * SEQ;

    bf16x16 aq0 = load_a_frag(Qh, q0,  0, HDIM);
    bf16x16 aq1 = load_a_frag(Qh, q0, 32, HDIM);

    f32x8 o0 = {}, o1 = {}, o2 = {}, o3 = {};
    float mrow[8], lrow[8];
    #pragma unroll
    for (int r = 0; r < 8; r++) { mrow[r] = -1e30f; lrow[r] = 0.f; }

    int n = lane & 15, half = lane >> 4;
    float slope = exp2f(-0.5f * (float)h);
    int kstart = q0 - WIN; if (kstart < 0) kstart = 0; kstart &= ~31;

    for (int kb = kstart; kb <= q0 + 15; kb += 32) {
        f32x8 s0 = {}, s1 = {};
        s0 = wmma_bf16(aq0, load_b_frag(Kh, kb,       0, HDIM), s0);
        s0 = wmma_bf16(aq1, load_b_frag(Kh, kb,      32, HDIM), s0);
        s1 = wmma_bf16(aq0, load_b_frag(Kh, kb + 16,  0, HDIM), s1);
        s1 = wmma_bf16(aq1, load_b_frag(Kh, kb + 16, 32, HDIM), s1);

        float p0[8], p1[8], v0[8], v1[8], nm[8];
        #pragma unroll
        for (int r = 0; r < 8; r++) {
            int i  = q0 + r + 8*half;
            int j0 = kb + n, j1 = kb + 16 + n;
            bool ok0 = (j0 <= i) && (i - j0 <= WIN);
            bool ok1 = (j1 <= i) && (i - j1 <= WIN);
            float x0 = ok0 ? (s0[r] * 0.125f + slope * (float)(j0 - i)) : -1e30f;
            float x1 = ok1 ? (s1[r] * 0.125f + slope * (float)(j1 - i)) : -1e30f;
            p0[r] = x0; p1[r] = x1;
            v0[r] = ok0 ? 1.f : 0.f; v1[r] = ok1 ? 1.f : 0.f;
            nm[r] = fmaxf(x0, x1);
        }
        #pragma unroll
        for (int mask = 1; mask < 16; mask <<= 1)
            #pragma unroll
            for (int r = 0; r < 8; r++) nm[r] = fmaxf(nm[r], __shfl_xor(nm[r], mask, 32));

        float alpha[8], rs[8];
        #pragma unroll
        for (int r = 0; r < 8; r++) {
            float newm = fmaxf(mrow[r], nm[r]);
            alpha[r] = __expf(mrow[r] - newm);
            mrow[r]  = newm;
            p0[r] = v0[r] * __expf(p0[r] - newm);
            p1[r] = v1[r] * __expf(p1[r] - newm);
            rs[r] = p0[r] + p1[r];
        }
        #pragma unroll
        for (int mask = 1; mask < 16; mask <<= 1)
            #pragma unroll
            for (int r = 0; r < 8; r++) rs[r] += __shfl_xor(rs[r], mask, 32);
        #pragma unroll
        for (int r = 0; r < 8; r++) {
            lrow[r] = lrow[r] * alpha[r] + rs[r];
            o0[r] *= alpha[r]; o1[r] *= alpha[r]; o2[r] *= alpha[r]; o3[r] *= alpha[r];
        }

        // C-layout -> A-layout for P via per-wave LDS bounce (bf16).
        __bf16* pw = &plds[wid][0];
        #pragma unroll
        for (int r = 0; r < 8; r++) {
            pw[(r + 8*half) * 32 +      n] = (__bf16)p0[r];
            pw[(r + 8*half) * 32 + 16 + n] = (__bf16)p1[r];
        }
        asm volatile("s_wait_dscnt 0" ::: "memory");
        const __bf16* pr = pw + (lane & 15) * 32;
        bf16x8 lo = *(const bf16x8*)(pr + half * 8);
        bf16x8 hi = *(const bf16x8*)(pr + 16 + half * 8);
        bf16x16 ap = __builtin_shufflevector(lo, hi, 0,1,2,3,4,5,6,7,8,9,10,11,12,13,14,15);

        o0 = wmma_bf16(ap, load_b_frag(Vh,  0, kb, SEQ), o0);
        o1 = wmma_bf16(ap, load_b_frag(Vh, 16, kb, SEQ), o1);
        o2 = wmma_bf16(ap, load_b_frag(Vh, 32, kb, SEQ), o2);
        o3 = wmma_bf16(ap, load_b_frag(Vh, 48, kb, SEQ), o3);
    }

    #pragma unroll
    for (int r = 0; r < 8; r++) {
        float inv = 1.0f / lrow[r];
        int i = q0 + r + 8*half;
        size_t ro = ((size_t)b * SEQ + i) * (HEADS*HDIM) + h * HDIM;
        Ob[ro +  0 + n] = (__bf16)(o0[r] * inv);
        Ob[ro + 16 + n] = (__bf16)(o1[r] * inv);
        Ob[ro + 32 + n] = (__bf16)(o2[r] * inv);
        Ob[ro + 48 + n] = (__bf16)(o3[r] * inv);
    }
}

// ---------------- launcher ----------------
extern "C" void kernel_launch(void* const* d_in, const int* in_sizes, int n_in,
                              void* d_out, int out_size, void* d_ws, size_t ws_size,
                              hipStream_t stream) {
    const float* features = (const float*)d_in[0];
    const float* wq = (const float*)d_in[1];
    const float* wk = (const float*)d_in[2];
    const float* wv = (const float*)d_in[3];
    const float* wo = (const float*)d_in[4];
    const float* qw = (const float*)d_in[5];
    const float* kw = (const float*)d_in[6];
    float* out = (float*)d_out;

    char* ws = (char*)d_ws;
    size_t off = 0;
    __bf16* Xbf  = (__bf16*)(ws + off); off += (size_t)ROWS * MODEL * 2;           // 8 MB
    __bf16* Wqkv = (__bf16*)(ws + off); off += (size_t)QKVN * MODEL * 2;           // 3 MB
    __bf16* Wo   = (__bf16*)(ws + off); off += (size_t)MODEL * (HEADS*HDIM) * 2;   // 2 MB
    float*  QKV  = (float*) (ws + off); off += (size_t)ROWS * QKVN * 4;            // 24 MB
    __bf16* Kbf  = (__bf16*)(ws + off); off += (size_t)BATCH*KVH*SEQ*HDIM * 2;     // 2 MB
    __bf16* Vt   = (__bf16*)(ws + off); off += (size_t)BATCH*KVH*HDIM*SEQ * 2;     // 2 MB
    __bf16* Qbf  = Xbf;          // alias: Xbf dead after QKV GEMM
    __bf16* Obf  = (__bf16*)QKV; // alias: QKV f32 dead after norm

    auto cvt = [&](const float* s, __bf16* d, int nElem) {
        cvt_f32_bf16<<<(nElem + 255) / 256, 256, 0, stream>>>(s, d, nElem);
    };
    cvt(features, Xbf, ROWS * MODEL);
    cvt(wq, Wqkv,                                           HEADS*HDIM * MODEL);
    cvt(wk, Wqkv + (size_t)(HEADS*HDIM) * MODEL,            KVH*HDIM * MODEL);
    cvt(wv, Wqkv + (size_t)(HEADS*HDIM + KVH*HDIM) * MODEL, KVH*HDIM * MODEL);
    cvt(wo, Wo, MODEL * HEADS*HDIM);

    // QKV projection: [4096,1536] = X[4096,1024] * Wqkv[1536,1024]^T
    gemm_bf16<<<dim3(QKVN / BN, ROWS / BM), 256, 0, stream>>>(Xbf, Wqkv, QKV, QKVN, MODEL);

    // RMSNorm + relayout
    norm_relayout<<<ROWS, 256, 0, stream>>>(QKV, qw, kw, Qbf, Kbf, Vt);

    // attention
    attn_kernel<<<(BATCH*HEADS*(SEQ/16)) / 8, 256, 0, stream>>>(Qbf, Kbf, Vt, Obf);

    // output projection: [4096,1024] = O[4096,1024] * Wo[1024,1024]^T
    gemm_bf16<<<dim3((HEADS*HDIM) / BN, ROWS / BM), 256, 0, stream>>>(Obf, Wo, out, MODEL, HEADS*HDIM);
}
